// TimeSeriesWeighting_498216206722
// MI455X (gfx1250) — matrix operand order repaired and checked by reading, not compiled, not run on hardware
//
#include <hip/hip_runtime.h>

// ---------------------------------------------------------------------------
// TimeSeriesWeighting on MI455X (gfx1250, wave32)
//
//   K0: Xh[1024][2528] f16 = x[:,0,:] zero-padded            (5.18 MB)
//   K1: Wt[2560][2528] f16 = DFT basis, N-major              (12.9 MB)
//   K2: WMMA GEMM with TDM (tensor_load_to_lds) double-buffered LDS staging;
//       power[k][b] = C[b,2k]^2 + C[b,2k+1]^2 -> powT[1280][1024]
//   K3: per-row top-3 pairs -> virtual top-4 -> patch hits
// ---------------------------------------------------------------------------

typedef __attribute__((ext_vector_type(16))) _Float16 v16h;
typedef __attribute__((ext_vector_type(8)))  _Float16 v8h;
typedef __attribute__((ext_vector_type(8)))  float    v8f;
typedef unsigned int v4u __attribute__((ext_vector_type(4)));
typedef int          v8i __attribute__((ext_vector_type(8)));
typedef int          v4i __attribute__((ext_vector_type(4)));

#define L_SIG   2500
#define K_PAD   2528               // 79 * 32
#define NT_K    79
#define N_BINS  1251               // bins 0..1250
#define N_COLS  2502               // 2 * N_BINS (cos/sin interleaved)
#define N_PAD   2560               // 20 * 128
#define M_ROWS  1024
#define M_BLK   128
#define N_BLK   128

__device__ __forceinline__ v16h mk16(v8h lo, v8h hi) {
  return __builtin_shufflevector(lo, hi, 0,1,2,3,4,5,6,7,8,9,10,11,12,13,14,15);
}

// ---- TDM: async 2D tile load (128 rows x 32 halves, row stride K_PAD) ------
// D# built per CDNA5 ISA (2-D tensor, groups 2/3
// zero).  clang-23 6-arg builtin form: (g0, g1, g2, g3, extra, cpol).
__device__ __forceinline__ void tdm_load_tile(unsigned lds_off, const void* gptr) {
  unsigned long long ga = (unsigned long long)gptr;
  v4u g0;
  g0[0] = 1u;                                        // count=1, user descriptor
  g0[1] = lds_off;                                   // lds_addr (bytes)
  g0[2] = (unsigned)(ga & 0xffffffffu);              // global_addr[31:0]
  g0[3] = (unsigned)((ga >> 32) & 0x01ffffffu)       // global_addr[56:32]
        | 0x80000000u;                               // type = 2 ("image")
  v8i g1;
  g1[0] = (int)(1u << 16);                           // data_size = 2 bytes
  g1[1] = (int)(32u  << 16);                         // tensor_dim0 = 32
  g1[2] = (int)(128u << 16);                         // tensor_dim1 = 128
  g1[3] = (int)(32u  << 16);                         // tile_dim0 = 32
  g1[4] = 128;                                       // tile_dim1=128, tile_dim2=0
  g1[5] = K_PAD;                                     // tensor_dim0_stride lo32
  g1[6] = 0;                                         // stride0 hi16 | stride1 lo16
  g1[7] = 0;
  v4i z4 = {0, 0, 0, 0};
  v8i z8 = {0, 0, 0, 0, 0, 0, 0, 0};
  __builtin_amdgcn_tensor_load_to_lds(g0, g1, z4, z4, z8, 0);
}

// ---- K0: convert x[:,0,:] to f16, zero-pad K to 2528 ------------------------
__global__ void build_xh(const float* __restrict__ x, _Float16* __restrict__ Xh) {
  int idx = blockIdx.x * blockDim.x + threadIdx.x;
  if (idx >= M_ROWS * K_PAD) return;
  int b = idx / K_PAD, k = idx - b * K_PAD;
  float v = (k < L_SIG) ? x[(size_t)b * (12 * L_SIG) + k] : 0.0f;
  Xh[idx] = (_Float16)v;
}

// ---- K1: DFT basis, stored N-major: Wt[col][k] ------------------------------
__global__ void build_wt(_Float16* __restrict__ Wt) {
  int idx = blockIdx.x * blockDim.x + threadIdx.x;
  if (idx >= N_PAD * K_PAD) return;
  int c = idx / K_PAD, n = idx - c * K_PAD;
  float v = 0.0f;
  if (n < L_SIG && c < N_COLS) {
    int k = c >> 1;
    int m = (k * n) % L_SIG;                       // exact in int32
    float th = (float)m * 2.5132741228718345e-3f;  // 2*pi/2500
    v = (c & 1) ? sinf(th) : cosf(th);
  }
  Wt[idx] = (_Float16)v;
}

// ---- K2: TDM-staged, double-buffered WMMA GEMM ------------------------------
__global__ __launch_bounds__(256) void dft_gemm(const _Float16* __restrict__ Xh,
                                                const _Float16* __restrict__ Wt,
                                                float* __restrict__ powT) {
  __shared__ _Float16 bufA[2][M_BLK * 32];   // 2 x 8 KB, packed [row][32]
  __shared__ _Float16 bufB[2][N_BLK * 32];   // 2 x 8 KB, packed [col][32]

  const int tid  = threadIdx.x;
  const int lane = tid & 31;
  const int w    = tid >> 5;                 // 8 waves
  const int blockN = blockIdx.x;             // 20
  const int blockM = blockIdx.y;             // 8

  const int mSub = (w & 3) * 2;              // 2 M-tiles per wave
  const int nSub = (w >> 2) * 4;             // 4 N-tiles per wave
  const int rsel  = lane & 15;
  const int aKoff = (lane & 16) ? 8 : 0;     // A: lanes16-31 hold K=8..15,24..31
  const int bKoff = (lane & 16) ? 8 : 0;     // B: lanes16-31 hold K=8..15,24..31

  const _Float16* gA = Xh + (size_t)(blockM * M_BLK) * K_PAD;
  const _Float16* gB = Wt + (size_t)(blockN * N_BLK) * K_PAD;

  v8f acc[2][4] = {};

  if (tid < 32) {                            // producer wave: preamble stage
    tdm_load_tile((unsigned)(unsigned long long)&bufA[0][0], gA);
    tdm_load_tile((unsigned)(unsigned long long)&bufB[0][0], gB);
  }

  for (int kt = 0; kt < NT_K; ++kt) {
    const int cur = kt & 1;
    if (tid < 32) {
      if (kt + 1 < NT_K) {                   // stage next k-panel, then wait cur
        const int nxt = cur ^ 1;
        tdm_load_tile((unsigned)(unsigned long long)&bufA[nxt][0],
                      gA + (size_t)(kt + 1) * 32);
        tdm_load_tile((unsigned)(unsigned long long)&bufB[nxt][0],
                      gB + (size_t)(kt + 1) * 32);
        __builtin_amdgcn_s_wait_tensorcnt(2);  // in-order: cur's pair done
      } else {
        __builtin_amdgcn_s_wait_tensorcnt(0);
      }
    }
    __syncthreads();                         // cur buffers visible to all waves

    v16h a[2], b[4];
#pragma unroll
    for (int i = 0; i < 2; ++i) {
      const _Float16* p = &bufA[cur][((mSub + i) * 16 + rsel) * 32 + aKoff];
      a[i] = mk16(*(const v8h*)p, *(const v8h*)(p + 16));
    }
#pragma unroll
    for (int j = 0; j < 4; ++j) {
      const _Float16* p = &bufB[cur][((nSub + j) * 16 + rsel) * 32 + bKoff];
      b[j] = mk16(*(const v8h*)p, *(const v8h*)(p + 16));
    }
#pragma unroll
    for (int i = 0; i < 2; ++i)
#pragma unroll
      for (int j = 0; j < 4; ++j)
        acc[i][j] = __builtin_amdgcn_wmma_f32_16x16x32_f16(
            false, a[i], false, b[j], (short)0, acc[i][j], false, false);

    __syncthreads();                         // cur free before kt+1 overwrites it
  }

  // Epilogue: power = cos^2 + sin^2 (even/odd lane pair), bin-major store.
  const int rowHalf = (lane & 16) ? 8 : 0;
  const bool writer = ((lane & 1) == 0);
#pragma unroll
  for (int i = 0; i < 2; ++i) {
    const int rowBase = blockM * M_BLK + (mSub + i) * 16 + rowHalf;
#pragma unroll
    for (int j = 0; j < 4; ++j) {
      const int bin = blockN * (N_BLK / 2) + (nSub + j) * 8 + (rsel >> 1);
#pragma unroll
      for (int r = 0; r < 8; ++r) {
        float p = acc[i][j][r] * acc[i][j][r];
        p += __shfl_xor(p, 1, 32);
        if (writer) powT[(size_t)bin * M_ROWS + rowBase + r] = p;
      }
    }
  }
}

// ---- K3: top-3 pairs -> virtual top-4 -> patch hits -------------------------
__global__ void topk_patches(const float* __restrict__ powT,
                             const float* __restrict__ wptr,
                             float* __restrict__ out) {
  int b = blockIdx.x * blockDim.x + threadIdx.x;
  if (b >= M_ROWS) return;

  float v0 = -1.f, v1 = -1.f, v2 = -1.f;
  int   i0 = 0,    i1 = 0,    i2 = 0;
  for (int k = 0; k < N_BINS; ++k) {            // coalesced: bin-major layout
    float v = powT[(size_t)k * M_ROWS + b];
    if (v > v0)      { v2 = v1; i2 = i1; v1 = v0; i1 = i0; v0 = v; i0 = k; }
    else if (v > v1) { v2 = v1; i2 = i1; v1 = v;  i1 = k; }
    else if (v > v2) { v2 = v;  i2 = k; }
  }

  int ids[3] = {i0, i1, i2};
  int e[6]; int ne = 0;
#pragma unroll
  for (int t = 0; t < 3; ++t) {
    int k = ids[t];
    e[ne++] = k;
    if (k != 0 && k != (L_SIG / 2)) e[ne++] = L_SIG - k;   // conjugate partner
  }
  const int s0 = e[0], s1 = e[2];               // idx[:, 0, ::2]

  int ks0 = (s0 <= (L_SIG - 1) / 2) ? s0 : s0 - L_SIG;
  int ks1 = (s1 <= (L_SIG - 1) / 2) ? s1 : s1 - L_SIG;
  int a0 = ks0 < 0 ? -ks0 : ks0;
  int a1 = ks1 < 0 ? -ks1 : ks1;
  bool val0 = (ks0 != 0), val1 = (ks1 != 0);
  int per0 = L_SIG / (a0 ? a0 : 1);
  int per1 = L_SIG / (a1 ? a1 : 1);

  const float wf = 12.0f * wptr[0];
#pragma unroll
  for (int p = 0; p < 50; ++p) {
    int start = p * 50;
    int c = 0;
    if (val0) { int fm = ((start + per0 - 1) / per0) * per0; if (fm < start + 50) ++c; }
    if (val1) { int fm = ((start + per1 - 1) / per1) * per1; if (fm < start + 50) ++c; }
    out[(size_t)b * 50 + p] = wf * (float)c;
  }
}

// ---------------------------------------------------------------------------
extern "C" void kernel_launch(void* const* d_in, const int* in_sizes, int n_in,
                              void* d_out, int out_size, void* d_ws, size_t ws_size,
                              hipStream_t stream) {
  const float* x = (const float*)d_in[0];
  const float* w = (const float*)d_in[1];
  float* out = (float*)d_out;

  char* ws = (char*)d_ws;
  const size_t XH_BYTES = (size_t)M_ROWS * K_PAD * 2;   //  5,177,344
  const size_t WT_BYTES = (size_t)N_PAD  * K_PAD * 2;   // 12,943,360
  _Float16* Xh   = (_Float16*)ws;
  _Float16* Wt   = (_Float16*)(ws + XH_BYTES);
  float*    powT = (float*)   (ws + XH_BYTES + WT_BYTES);  // 1280*1024*4

  {
    int n = M_ROWS * K_PAD;
    build_xh<<<(n + 255) / 256, 256, 0, stream>>>(x, Xh);
  }
  {
    int n = N_PAD * K_PAD;
    build_wt<<<(n + 255) / 256, 256, 0, stream>>>(Wt);
  }
  dft_gemm<<<dim3(N_PAD / N_BLK, M_ROWS / M_BLK), 256, 0, stream>>>(Xh, Wt, powT);
  topk_patches<<<(M_ROWS + 255) / 256, 256, 0, stream>>>(powT, w, out);
}
